// HybridCameraBEVModel_29566554866351
// MI455X (gfx1250) — matrix operand order repaired and checked by reading, not compiled, not run on hardware
//
#include <hip/hip_runtime.h>
#include <hip/hip_bf16.h>

typedef __attribute__((ext_vector_type(16))) __bf16 v16bf;
typedef __attribute__((ext_vector_type(8)))  float  v8f;

union FragBF16 { v16bf v; unsigned int u[8]; };

__device__ __forceinline__ unsigned int pack2_bf16(float a, float b) {
    unsigned int ua = __float_as_uint(a);
    unsigned int ub = __float_as_uint(b);
    ua += 0x7fffu + ((ua >> 16) & 1u);   // round-to-nearest-even
    ub += 0x7fffu + ((ub >> 16) & 1u);
    return (ua >> 16) | (ub & 0xffff0000u);
}
__device__ __forceinline__ unsigned short to_bf16(float a) {
    unsigned int ua = __float_as_uint(a);
    ua += 0x7fffu + ((ua >> 16) & 1u);
    return (unsigned short)(ua >> 16);
}

// ---------------------------------------------------------------------------
// C[M,N] = A[M,256] * B[256,N] + bias[N]   (fp32 in/out, bf16 WMMA compute)
// Block: 256 threads (8 waves). Block tile 128x128, K-step 32.
// Wave grid 2(M) x 4(N); wave tile 64x32 -> 4x2 fragments of 16x16.
// Software-pipelined: K-step k+1's global tiles are fetched into registers
// while K-step k's WMMAs consume LDS.
// ---------------------------------------------------------------------------
__global__ __launch_bounds__(256) void gemm_bf16_wmma(
    const float* __restrict__ A, const float* __restrict__ B,
    const float* __restrict__ bias, float* __restrict__ C,
    int M, int N)
{
    const int K = 256;
    __shared__ unsigned int As[128][17];                  // [m][k-pair], padded
    __shared__ __align__(16) unsigned short Bs[128][34];  // [n][k], padded (17 uints)

    const int t    = threadIdx.x;
    const int wave = t >> 5;
    const int lane = t & 31;
    const int lo   = lane & 15;
    const int hi   = lane >> 4;
    const int waveM = wave >> 2;   // 0..1
    const int waveN = wave & 3;    // 0..3
    const int rowBlock = blockIdx.x * 128;
    const int colBlock = blockIdx.y * 128;

    // Per-thread staging coordinates (loop-invariant).
    int arow[4], ac4[4], bkr[4], bn4[4];
    bool aval[4];
#pragma unroll
    for (int i = 0; i < 4; ++i) {
        int idx = t + i * 256;           // 0..1023 float4 slots
        arow[i] = idx >> 3;              // A: row 0..127
        ac4[i]  = idx & 7;               // A: float4 within 32-col row
        aval[i] = (rowBlock + arow[i]) < M;
        bkr[i]  = idx >> 5;              // B: k-row 0..31
        bn4[i]  = idx & 31;              // B: float4 within 128-col row
    }

    float4 ra[4], rb[4];
    auto load_stage = [&](int k0) {
#pragma unroll
        for (int i = 0; i < 4; ++i) {
            ra[i] = make_float4(0.f, 0.f, 0.f, 0.f);
            if (aval[i])
                ra[i] = *(const float4*)(A + (size_t)(rowBlock + arow[i]) * K + k0 + ac4[i] * 4);
            rb[i] = *(const float4*)(B + (size_t)(k0 + bkr[i]) * N + colBlock + bn4[i] * 4);
        }
    };
    auto store_stage = [&]() {
#pragma unroll
        for (int i = 0; i < 4; ++i) {
            As[arow[i]][ac4[i] * 2 + 0] = pack2_bf16(ra[i].x, ra[i].y);
            As[arow[i]][ac4[i] * 2 + 1] = pack2_bf16(ra[i].z, ra[i].w);
            Bs[bn4[i] * 4 + 0][bkr[i]] = to_bf16(rb[i].x);
            Bs[bn4[i] * 4 + 1][bkr[i]] = to_bf16(rb[i].y);
            Bs[bn4[i] * 4 + 2][bkr[i]] = to_bf16(rb[i].z);
            Bs[bn4[i] * 4 + 3][bkr[i]] = to_bf16(rb[i].w);
        }
    };

    v8f acc[4][2];
#pragma unroll
    for (int tm = 0; tm < 4; ++tm)
#pragma unroll
        for (int tn = 0; tn < 2; ++tn)
            acc[tm][tn] = (v8f){0.f,0.f,0.f,0.f,0.f,0.f,0.f,0.f};

    load_stage(0);                       // prologue fetch
    for (int k0 = 0; k0 < K; k0 += 32) {
        store_stage();
        __syncthreads();
        if (k0 + 32 < K)
            load_stage(k0 + 32);         // overlap next fetch with this step's math

        // ---- load fragments per ISA 7.12.2 16-bit layouts ----
        FragBF16 afr[4], bfr[2];
#pragma unroll
        for (int tm = 0; tm < 4; ++tm) {
            int m = waveM * 64 + tm * 16 + lo;
#pragma unroll
            for (int j = 0; j < 8; ++j)   // k-pair = (j&3) + (j>>2)*8 + hi*4
                afr[tm].u[j] = As[m][(j & 3) + ((j >> 2) << 3) + hi * 4];
        }
#pragma unroll
        for (int tn = 0; tn < 2; ++tn) {
            int n = waveN * 32 + tn * 16 + lo;
            const unsigned int* brow = (const unsigned int*)&Bs[n][0];
#pragma unroll
            for (int j = 0; j < 8; ++j)   // k-pair = (j&3) + (j>>2)*8 + hi*4
                bfr[tn].u[j] = brow[(j & 3) + ((j >> 2) << 3) + hi * 4];
        }
#pragma unroll
        for (int tm = 0; tm < 4; ++tm)
#pragma unroll
            for (int tn = 0; tn < 2; ++tn)
                acc[tm][tn] = __builtin_amdgcn_wmma_f32_16x16x32_bf16(
                    false, afr[tm].v, false, bfr[tn].v,
                    (short)0, acc[tm][tn], false, false);
        __syncthreads();                 // LDS safe to overwrite next iteration
    }

    // ---- epilogue: D layout lane lo=N, VGPR r -> m = r + hi*8 ----
#pragma unroll
    for (int tn = 0; tn < 2; ++tn) {
        int col = colBlock + waveN * 32 + tn * 16 + lo;
        float bv = bias[col];
#pragma unroll
        for (int tm = 0; tm < 4; ++tm) {
#pragma unroll
            for (int r = 0; r < 8; ++r) {
                int row = rowBlock + waveM * 64 + tm * 16 + r + hi * 8;
                if (row < M)
                    C[(size_t)row * N + col] = acc[tm][tn][r] + bv;
            }
        }
    }
}

// ---------------------------------------------------------------------------
// Deformable sampling: one wave per (b, h, q); lane = channel c (d = 32).
// Lane s also owns sample s's softmax weight and grid coords.
// ---------------------------------------------------------------------------
__global__ __launch_bounds__(256) void msdeform_sample(
    const float* __restrict__ v,     // [bs*Ltot, 256]  (= [b][s][h][32])
    const float* __restrict__ off,   // [bs*Q, 512]
    const float* __restrict__ attn,  // [bs*Q, 256]
    const float* __restrict__ ref,   // [bs*Q, 4, 2]
    float* __restrict__ out)         // [bs*Q, 256]
{
    const int Q = 4096, Ltot = 19560;
    const int Wl_[4] = {160, 80, 40, 20};
    const int Hl_[4] = {92, 46, 23, 12};
    const int st_[4] = {0, 14720, 18400, 19320};

    int wid  = blockIdx.x * 8 + (threadIdx.x >> 5);
    int lane = threadIdx.x & 31;
    int q = wid & (Q - 1);
    int h = (wid >> 12) & 7;
    int b = wid >> 15;
    int row = b * Q + q;

    // softmax over the 32 attention logits (lane i holds logit i)
    float lg = attn[(size_t)row * 256 + h * 32 + lane];
    float mx = lg;
#pragma unroll
    for (int o = 16; o > 0; o >>= 1) mx = fmaxf(mx, __shfl_xor(mx, o, 32));
    float e = __expf(lg - mx);
    float sm = e;
#pragma unroll
    for (int o = 16; o > 0; o >>= 1) sm += __shfl_xor(sm, o, 32);
    float aw = e / sm;

    // lane s = lv*8+p: pixel coords of sample s
    int   lv = lane >> 3;
    float ox = off[(size_t)row * 512 + h * 64 + lane * 2 + 0];
    float oy = off[(size_t)row * 512 + h * 64 + lane * 2 + 1];
    float rx = ref[((size_t)row * 4 + lv) * 2 + 0];
    float ry = ref[((size_t)row * 4 + lv) * 2 + 1];
    float xs = rx * (float)Wl_[lv] + ox - 0.5f;   // = loc_x*W - 0.5
    float ys = ry * (float)Hl_[lv] + oy - 0.5f;

    const float* vb = v + (size_t)b * Ltot * 256 + h * 32 + lane;
    float accv = 0.f;
#pragma unroll
    for (int lvl = 0; lvl < 4; ++lvl) {
        const int Wl = Wl_[lvl], Hl = Hl_[lvl];
        const float* vlev = vb + (size_t)st_[lvl] * 256;
#pragma unroll
        for (int p = 0; p < 8; ++p) {
            int s = lvl * 8 + p;
            float xf = __shfl(xs, s, 32);   // uniform across wave
            float yf = __shfl(ys, s, 32);
            float w  = __shfl(aw, s, 32);
            float x0f = floorf(xf), y0f = floorf(yf);
            float wx = xf - x0f, wy = yf - y0f;
            int x0 = (int)x0f, y0 = (int)y0f;
            bool vx0 = (x0 >= 0) && (x0 < Wl);
            bool vx1 = (x0 + 1 >= 0) && (x0 + 1 < Wl);
            bool vy0 = (y0 >= 0) && (y0 < Hl);
            bool vy1 = (y0 + 1 >= 0) && (y0 + 1 < Hl);
            float v00 = 0.f, v01 = 0.f, v10 = 0.f, v11 = 0.f;
            if (vy0) {
                const float* r0 = vlev + (size_t)(y0 * Wl) * 256;
                if (vx0) v00 = r0[(size_t)x0 * 256];
                if (vx1) v01 = r0[(size_t)(x0 + 1) * 256];
            }
            if (vy1) {
                const float* r1 = vlev + (size_t)((y0 + 1) * Wl) * 256;
                if (vx0) v10 = r1[(size_t)x0 * 256];
                if (vx1) v11 = r1[(size_t)(x0 + 1) * 256];
            }
            accv += w * ((1.f - wx) * (1.f - wy) * v00 + wx * (1.f - wy) * v01 +
                         (1.f - wx) * wy * v10 + wx * wy * v11);
        }
    }
    out[(size_t)row * 256 + h * 32 + lane] = accv;
}

extern "C" void kernel_launch(void* const* d_in, const int* in_sizes, int n_in,
                              void* d_out, int out_size, void* d_ws, size_t ws_size,
                              hipStream_t stream) {
    (void)in_sizes; (void)n_in; (void)out_size; (void)ws_size;
    const float* query  = (const float*)d_in[0];
    const float* value  = (const float*)d_in[1];
    const float* refpts = (const float*)d_in[2];
    /* d_in[3]: spatial_shapes (known constants, baked in) */
    const float* W_off  = (const float*)d_in[4];
    const float* b_off  = (const float*)d_in[5];
    const float* W_attn = (const float*)d_in[6];
    const float* b_attn = (const float*)d_in[7];
    const float* W_val  = (const float*)d_in[8];
    const float* b_val  = (const float*)d_in[9];
    const float* W_out  = (const float*)d_in[10];
    const float* b_out  = (const float*)d_in[11];
    float* out = (float*)d_out;

    const int bs = 6, Q = 4096, Ltot = 19560;
    const int Mv = bs * Ltot;   // 117360
    const int Mq = bs * Q;      // 24576

    float* ws      = (float*)d_ws;
    float* v_ws    = ws;                          // Mv*256
    float* off_ws  = v_ws   + (size_t)Mv * 256;   // Mq*512
    float* attn_ws = off_ws + (size_t)Mq * 512;   // Mq*256
    float* samp_ws = attn_ws + (size_t)Mq * 256;  // Mq*256

    dim3 blk(256);
    // v = value @ W_val + b_val        (layout already [b][s][h][d])
    gemm_bf16_wmma<<<dim3((Mv + 127) / 128, 2), blk, 0, stream>>>(
        value, W_val, b_val, v_ws, Mv, 256);
    // off = query @ W_off + b_off
    gemm_bf16_wmma<<<dim3((Mq + 127) / 128, 4), blk, 0, stream>>>(
        query, W_off, b_off, off_ws, Mq, 512);
    // attn logits = query @ W_attn + b_attn
    gemm_bf16_wmma<<<dim3((Mq + 127) / 128, 2), blk, 0, stream>>>(
        query, W_attn, b_attn, attn_ws, Mq, 256);
    // softmax + bilinear gather + weighted sum -> [b][q][h][d]
    msdeform_sample<<<dim3(bs * 8 * Q / 8), blk, 0, stream>>>(
        v_ws, off_ws, attn_ws, refpts, samp_ws);
    // out = sampled @ W_out + b_out
    gemm_bf16_wmma<<<dim3((Mq + 127) / 128, 2), blk, 0, stream>>>(
        samp_ws, W_out, b_out, out, Mq, 256);
}
